// Generator_76312978915883
// MI455X (gfx1250) — compile-verified, matrix operand order checked by reading, not run on hardware
//
#include <hip/hip_runtime.h>
#include <hip/hip_bf16.h>

typedef __bf16 bf16;
typedef __attribute__((ext_vector_type(16))) __bf16 v16bf;
typedef __attribute__((ext_vector_type(8)))  __bf16 v8bf;
typedef __attribute__((ext_vector_type(8)))  float   v8f;

#define BB 64
#define TT 100
#define VV 16000
#define EE 512
#define HH 1024
#define G4H 4096
#define START_TOKEN 2

// LDS B-tile: 64 columns x 32 K, padded column stride of 40 elements (80 B)
// -> fragment reads are bank-conflict-free b128s on the 64-bank LDS.
#define BCOL_STRIDE 40

// ---------------------------------------------------------------- helpers
__device__ __forceinline__ float sigmoidf(float x) {
    return 1.0f / (1.0f + __expf(-x));
}

// ---------------------------------------------------------------- converts
__global__ void f32_to_bf16_k(const float* __restrict__ in, bf16* __restrict__ out, long n) {
    long i = (long)blockIdx.x * blockDim.x + threadIdx.x;
    long stride = (long)gridDim.x * blockDim.x;
    for (; i < n; i += stride) out[i] = (bf16)in[i];
}

__global__ void bias_sum_k(const float* __restrict__ b_ih, const float* __restrict__ b_hh,
                           float* __restrict__ bias_sum) {
    int i = blockIdx.x * blockDim.x + threadIdx.x;
    if (i < G4H) bias_sum[i] = b_ih[i] + b_hh[i];
}

__global__ void zero_state_k(float* __restrict__ c, bf16* __restrict__ h) {
    int i = blockIdx.x * blockDim.x + threadIdx.x;
    if (i < BB * HH) { c[i] = 0.0f; h[i] = (bf16)0.0f; }
}

// ------------------------------------------------- embedding gather+concat
__global__ void build_inp_k(const int* __restrict__ x, const int* __restrict__ cond,
                            const float* __restrict__ wt, const float* __restrict__ ct,
                            bf16* __restrict__ inp) {
    int bt = blockIdx.x;
    int b = bt / TT, t = bt % TT;
    int token = (t == 0) ? START_TOKEN : x[b * TT + t - 1];
    int cidx = cond[b];
    bf16* row = inp + (long)bt * (2 * EE);
    const float* wrow = wt + (long)token * EE;
    const float* crow = ct + (long)cidx * EE;
#pragma unroll
    for (int i = 0; i < 4; ++i) {
        int e = threadIdx.x + i * 256;           // coalesced
        float v = (e < EE) ? wrow[e] : crow[e - EE];
        row[e] = (bf16)v;
    }
}

// ---------------------------------------------------------------- WMMA GEMM
// C[M,N](f32) = A[M,K](bf16 row-major, lda) x B[N,K](bf16 row-major, ldb)^T
//               + bias[N] (optional) + addend[m*addStride + n] (optional)
//
// Block: 256 threads = 8 waves. Block tile: (128*ROWS)M x 64N; wave w owns a
// (16*ROWS) x 64 strip = ROWS x 4 WMMA accumulators. The 64x32 B tile is
// staged once per block per K-step into double-buffered LDS and shared by all
// 8 waves. Next K-step's A fragments + staging registers are fetched before
// the current WMMA burst; B fragments are preloaded into an array so the
// scheduler can use partial s_wait_dscnt values across the 4*ROWS WMMAs.
template <int ROWS>
__global__ __launch_bounds__(256)
void gemm_bf16_wmma(const bf16* __restrict__ A, long lda,
                    const bf16* __restrict__ B, long ldb,
                    float* __restrict__ C, long ldc,
                    int M, int N, int K,
                    const float* __restrict__ bias,
                    const float* __restrict__ addend, long addStride) {
    __shared__ __align__(16) bf16 Bsh[2][64 * BCOL_STRIDE];

    const int tid  = threadIdx.x;
    const int wave = tid >> 5;
    const int lane = tid & 31;
    const int hl   = lane >> 4;                  // lane half: 0 | 1
    const int l15  = lane & 15;
    const int koff = hl << 3;                    // 0 | 8

    const int m0 = blockIdx.y * (128 * ROWS) + wave * (16 * ROWS);
    const int n0 = blockIdx.x * 64;

    // A fragment pointers (row clamped so out-of-range strips read valid mem;
    // their stores are guarded below; all waves must reach the barriers)
    const bf16* aptr[ROWS];
#pragma unroll
    for (int i = 0; i < ROWS; ++i) {
        int rowA = m0 + 16 * i + l15;
        if (rowA >= M) rowA = M - 1;
        aptr[i] = A + (long)rowA * lda + koff;
    }

    // B staging: thread tid copies one 16B chunk per K-step.
    const int scol = tid >> 2;                   // 0..63
    const int skc  = (tid & 3) << 3;             // 0,8,16,24
    const bf16* sptr  = B + (long)(n0 + scol) * ldb + skc;
    bf16*       sdst0 = &Bsh[0][scol * BCOL_STRIDE + skc];
    bf16*       sdst1 = &Bsh[1][scol * BCOL_STRIDE + skc];

    v8f acc[ROWS][4];
#pragma unroll
    for (int i = 0; i < ROWS; ++i)
#pragma unroll
        for (int j = 0; j < 4; ++j) acc[i][j] = (v8f){};

    const int nk = K >> 5;                       // K/32 steps (K % 32 == 0 here)

    // prologue: stage k-step 0, preload A frags for step 0
    v8bf sreg = *(const v8bf*)(sptr);
    *(v8bf*)sdst0 = sreg;
    v8bf aLo[ROWS], aHi[ROWS];
#pragma unroll
    for (int i = 0; i < ROWS; ++i) {
        aLo[i] = *(const v8bf*)(aptr[i]);
        aHi[i] = *(const v8bf*)(aptr[i] + 16);
    }
    __syncthreads();

    for (int kt = 0; kt < nk; ++kt) {
        const int cur = kt & 1;
        const bool hasNext = (kt + 1) < nk;

        // issue next K-step's global loads first (overlap with WMMA burst)
        v8bf sregN, aLoN[ROWS], aHiN[ROWS];
        if (hasNext) {
            const long kn = (long)(kt + 1) << 5;
            sregN = *(const v8bf*)(sptr + kn);
#pragma unroll
            for (int i = 0; i < ROWS; ++i) {
                aLoN[i] = *(const v8bf*)(aptr[i] + kn);
                aHiN[i] = *(const v8bf*)(aptr[i] + kn + 16);
            }
        }

        // preload all 4 B fragments from LDS (distinct regs -> partial waits)
        v16bf bfr[4];
#pragma unroll
        for (int j = 0; j < 4; ++j) {
            const bf16* bp = &Bsh[cur][(j * 16 + l15) * BCOL_STRIDE + koff];
            v8bf b_lo = *(const v8bf*)(bp);
            v8bf b_hi = *(const v8bf*)(bp + 16);
            bfr[j] = __builtin_shufflevector(b_lo, b_hi,
                        0, 1, 2, 3, 4, 5, 6, 7, 8, 9, 10, 11, 12, 13, 14, 15);
        }

        v16bf af[ROWS];
#pragma unroll
        for (int i = 0; i < ROWS; ++i)
            af[i] = __builtin_shufflevector(aLo[i], aHi[i],
                        0, 1, 2, 3, 4, 5, 6, 7, 8, 9, 10, 11, 12, 13, 14, 15);

#pragma unroll
        for (int j = 0; j < 4; ++j)
#pragma unroll
            for (int i = 0; i < ROWS; ++i)
                acc[i][j] = __builtin_amdgcn_wmma_f32_16x16x32_bf16(
                                false, af[i], false, bfr[j], (short)0,
                                acc[i][j], false, false);

        if (hasNext) {
            __syncthreads();                     // buf[cur^1] reads (kt-1) done
            *(v8bf*)(cur ? sdst0 : sdst1) = sregN;
#pragma unroll
            for (int i = 0; i < ROWS; ++i) { aLo[i] = aLoN[i]; aHi[i] = aHiN[i]; }
            __syncthreads();                     // staged tile visible
        }
    }

    // D layout: lanes 0-15 -> M = vgpr idx, N = lane; lanes 16-31 -> M = 8+idx.
#pragma unroll
    for (int i = 0; i < ROWS; ++i) {
        const int mi = m0 + 16 * i;
        if (mi >= M) continue;                   // dead strip: no stores
        const int rbase = mi + (hl << 3);
#pragma unroll
        for (int j = 0; j < 4; ++j) {
            const int col = n0 + j * 16 + l15;
            const float badd = bias ? bias[col] : 0.0f;
#pragma unroll
            for (int r = 0; r < 8; ++r) {
                const long row = rbase + r;
                float v = acc[i][j][r] + badd;
                if (addend) v += addend[row * addStride + col];
                C[row * ldc + col] = v;
            }
        }
    }
}

// ---------------------------------------------------------------- LSTM gates
__global__ void lstm_gates_k(const float* __restrict__ g, float* __restrict__ c,
                             bf16* __restrict__ h_bf, bf16* __restrict__ hs_bf, int t) {
    int idx = blockIdx.x * blockDim.x + threadIdx.x;   // < B*H
    int b = idx >> 10, j = idx & (HH - 1);
    const float* gr = g + (long)b * G4H;
    float i  = sigmoidf(gr[j]);
    float f  = sigmoidf(gr[HH + j]);
    float gg = tanhf(gr[2 * HH + j]);
    float o  = sigmoidf(gr[3 * HH + j]);
    float cn = f * c[idx] + i * gg;
    float h  = o * tanhf(cn);
    c[idx] = cn;
    h_bf[idx] = (bf16)h;
    hs_bf[((long)b * TT + t) * HH + j] = (bf16)h;      // [B,T,H] layout
}

// ---------------------------------------------------------------- launcher
extern "C" void kernel_launch(void* const* d_in, const int* in_sizes, int n_in,
                              void* d_out, int out_size, void* d_ws, size_t ws_size,
                              hipStream_t stream) {
    const int*   x        = (const int*)d_in[0];
    const int*   cond     = (const int*)d_in[1];
    const float* word_tab = (const float*)d_in[2];
    const float* cond_tab = (const float*)d_in[3];
    const float* W_ih     = (const float*)d_in[4];
    const float* b_ih     = (const float*)d_in[5];
    const float* W_hh     = (const float*)d_in[6];
    const float* b_hh     = (const float*)d_in[7];
    const float* W_out    = (const float*)d_in[8];
    const float* b_out    = (const float*)d_in[9];
    float* out = (float*)d_out;

    // workspace partition (256B aligned)
    char* base = (char*)d_ws;
    size_t off = 0;
    auto take = [&](size_t bytes) -> char* {
        off = (off + 255) & ~(size_t)255;
        char* p = base + off;
        off += bytes;
        return p;
    };
    bf16*  W_ih_bf  = (bf16*)take((size_t)G4H * (2 * EE) * 2);
    bf16*  W_hh_bf  = (bf16*)take((size_t)G4H * HH * 2);
    bf16*  W_out_bf = (bf16*)take((size_t)VV * HH * 2);
    bf16*  inp_bf   = (bf16*)take((size_t)BB * TT * (2 * EE) * 2);
    bf16*  hs_bf    = (bf16*)take((size_t)BB * TT * HH * 2);
    float* gx       = (float*)take((size_t)BB * TT * G4H * 4);
    float* bias_s   = (float*)take((size_t)G4H * 4);
    float* gbuf     = (float*)take((size_t)BB * G4H * 4);
    float* c_st     = (float*)take((size_t)BB * HH * 4);
    bf16*  h_bf     = (bf16*)take((size_t)BB * HH * 2);
    (void)ws_size; (void)in_sizes; (void)n_in; (void)out_size;

    // weight conversions (one big L2-resident bf16 copy each)
    f32_to_bf16_k<<<2048, 256, 0, stream>>>(W_ih,  W_ih_bf,  (long)G4H * (2 * EE));
    f32_to_bf16_k<<<2048, 256, 0, stream>>>(W_hh,  W_hh_bf,  (long)G4H * HH);
    f32_to_bf16_k<<<4096, 256, 0, stream>>>(W_out, W_out_bf, (long)VV * HH);
    bias_sum_k<<<(G4H + 255) / 256, 256, 0, stream>>>(b_ih, b_hh, bias_s);
    zero_state_k<<<(BB * HH + 255) / 256, 256, 0, stream>>>(c_st, h_bf);

    // embedding gather + concat -> bf16 inp [B*T, 1024]
    build_inp_k<<<BB * TT, 256, 0, stream>>>(x, cond, word_tab, cond_tab, inp_bf);

    // gx[B*T,4H] = inp x W_ih^T + (b_ih + b_hh)   (256-row blocks, 2 rows/wave)
    {
        dim3 g(G4H / 64, (BB * TT) / 256);
        gemm_bf16_wmma<2><<<g, 256, 0, stream>>>(inp_bf, 2 * EE, W_ih_bf, 2 * EE,
                                                 gx, G4H, BB * TT, G4H, 2 * EE,
                                                 bias_s, nullptr, 0);
    }

    // sequential LSTM scan (M=64: ROWS=1, waves 4..7 compute but don't store)
    for (int t = 0; t < TT; ++t) {
        dim3 g(G4H / 64, 1);
        gemm_bf16_wmma<1><<<g, 256, 0, stream>>>(h_bf, HH, W_hh_bf, HH,
                                                 gbuf, G4H, BB, G4H, HH,
                                                 nullptr,
                                                 gx + (long)t * G4H, (long)TT * G4H);
        lstm_gates_k<<<(BB * HH) / 256, 256, 0, stream>>>(gbuf, c_st, h_bf, hs_bf, t);
    }

    // logits[B,T,V] = hs x W_out^T + b_out  (writes d_out directly)
    {
        dim3 g(VV / 64, (BB * TT) / 256);
        gemm_bf16_wmma<2><<<g, 256, 0, stream>>>(hs_bf, HH, W_out_bf, HH,
                                                 out, VV, BB * TT, VV, HH,
                                                 b_out, nullptr, 0);
    }
}
